// CharDecoder_80702435492319
// MI455X (gfx1250) — matrix operand order repaired and checked by reading, not compiled, not run on hardware
//
#include <hip/hip_runtime.h>
#include <stdint.h>

// ---------------------------------------------------------------------------
// Problem constants
// ---------------------------------------------------------------------------
#define LL 64
#define BB 2048
#define HH 512
#define EE 50
#define VV 96
#define EPAD 64                 // E padded to 64 for K alignment
#define KT  (EPAD + HH)         // 576 : fused K of [x | h]
#define XS  KT                  // xh row stride in elements (576 bf16 = 1152 B, 16B aligned)

typedef __attribute__((ext_vector_type(16))) __bf16 v16bf;
typedef __attribute__((ext_vector_type(8)))  float  v8f;
typedef __attribute__((ext_vector_type(4)))  unsigned int u32x4;

union AFrag { v16bf v; u32x4 q[2]; };

__device__ __forceinline__ unsigned short f2bf(float f) {
    union { float f; unsigned int u; } v; v.f = f;
    unsigned int u = v.u;
    u = (u + 0x7fffu + ((u >> 16) & 1u)) >> 16;   // round-to-nearest-even
    return (unsigned short)u;
}
__device__ __forceinline__ float bf2f(unsigned short s) {
    union { unsigned int u; float f; } v; v.u = ((unsigned int)s) << 16;
    return v.f;
}
__device__ __forceinline__ float sigf(float x) { return 1.0f / (1.0f + __expf(-x)); }

// ---------------------------------------------------------------------------
// Setup kernels (run once per launch; cheap)
// ---------------------------------------------------------------------------
__global__ void pack_wcat(const float* __restrict__ Wih, const float* __restrict__ Whh,
                          unsigned short* __restrict__ Wcat) {
    int idx = blockIdx.x * blockDim.x + threadIdx.x;          // 4H * KT
    if (idx >= 4 * HH * KT) return;
    int r = idx / KT, k = idx % KT;
    float v = 0.0f;
    if (k < EE)            v = Wih[r * EE + k];
    else if (k >= EPAD)    v = Whh[r * HH + (k - EPAD)];
    Wcat[idx] = f2bf(v);
}

__global__ void pack_emb(const float* __restrict__ emb, unsigned short* __restrict__ embp) {
    int idx = blockIdx.x * blockDim.x + threadIdx.x;          // V * EPAD
    if (idx >= VV * EPAD) return;
    int v = idx >> 6, e = idx & 63;
    embp[idx] = (e < EE) ? f2bf(emb[v * EE + e]) : (unsigned short)0;
}

__global__ void pack_wout(const float* __restrict__ W, unsigned short* __restrict__ Wb) {
    int idx = blockIdx.x * blockDim.x + threadIdx.x;          // V * H
    if (idx >= VV * HH) return;
    Wb[idx] = f2bf(W[idx]);
}

__global__ void pack_bias(const float* __restrict__ bih, const float* __restrict__ bhh,
                          float* __restrict__ bias) {
    int idx = blockIdx.x * blockDim.x + threadIdx.x;          // 4H
    if (idx >= 4 * HH) return;
    bias[idx] = bih[idx] + bhh[idx];
}

__global__ void init_state(const float* __restrict__ h0, const float* __restrict__ c0,
                           unsigned short* __restrict__ xhA, float* __restrict__ c) {
    int idx = blockIdx.x * blockDim.x + threadIdx.x;          // B * H
    if (idx >= BB * HH) return;
    int b = idx / HH, j = idx % HH;
    xhA[(size_t)b * XS + EPAD + j] = f2bf(h0[idx]);
    c[idx] = c0[idx];
}

// per-step embedding gather into x-part of the current xh buffer
__global__ void embed_x(const int* __restrict__ ids_t, const unsigned short* __restrict__ embp,
                        unsigned short* __restrict__ xh) {
    int idx = blockIdx.x * blockDim.x + threadIdx.x;          // B * EPAD
    if (idx >= BB * EPAD) return;
    int b = idx >> 6, e = idx & 63;
    xh[(size_t)b * XS + e] = embp[ids_t[b] * EPAD + e];
}

__global__ void finalize(const unsigned short* __restrict__ xh_fin, const float* __restrict__ c,
                         float* __restrict__ out_tail) {
    int idx = blockIdx.x * blockDim.x + threadIdx.x;          // B * H
    if (idx >= BB * HH) return;
    int b = idx / HH, j = idx % HH;
    out_tail[idx] = bf2f(xh_fin[(size_t)b * XS + EPAD + j]);  // hL
    out_tail[BB * HH + idx] = c[idx];                         // cL
}

// ---------------------------------------------------------------------------
// Fused LSTM step: gates = [x|h] @ Wcat^T + bias  ->  nonlinearity  ->
//                  c (f32, in place), h_new (bf16 into next xh buffer)
// grid (B/128, H/32), 128 threads (4 waves). Wave w: rows 32w..32w+31, and all
// four gate column groups {g*H + j0 .. +31 : g=0..3} -> 16 WMMA tiles / wave.
// ---------------------------------------------------------------------------
__global__ __launch_bounds__(128) void lstm_step(
        const unsigned short* __restrict__ xh,
        unsigned short* __restrict__ xh_next,
        const unsigned short* __restrict__ Wcat,
        const float* __restrict__ bias,
        float* __restrict__ c) {
    const int wave = threadIdx.x >> 5;
    const int lane = threadIdx.x & 31;
    const int half = lane >> 4;          // K-half selector per WMMA layout
    const int lr   = lane & 15;
    const int row0 = blockIdx.x * 128 + wave * 32;   // batch rows of this wave
    const int j0   = blockIdx.y * 32;                // hidden-unit column base

    v8f acc[16] = {};                                // [mt*8 + g*2 + s]

    const unsigned short* arow0 = xh + (size_t)(row0 + lr) * XS;
    const unsigned short* arow1 = arow0 + (size_t)16 * XS;

    for (int k = 0; k < KT; k += 32) {
        AFrag a0, a1;
        a0.q[0] = *(const u32x4*)(arow0 + k + half * 8);
        a0.q[1] = *(const u32x4*)(arow0 + k + 16 + half * 8);
        a1.q[0] = *(const u32x4*)(arow1 + k + half * 8);
        a1.q[1] = *(const u32x4*)(arow1 + k + 16 + half * 8);
#pragma unroll
        for (int n = 0; n < 8; n++) {                // n = g*2 + s
            const unsigned short* brow =
                Wcat + (size_t)((n >> 1) * HH + j0 + (n & 1) * 16 + lr) * KT + k + half * 8;
            AFrag b;
            b.q[0] = *(const u32x4*)(brow);
            b.q[1] = *(const u32x4*)(brow + 16);
            acc[n]     = __builtin_amdgcn_wmma_f32_16x16x32_bf16(
                             false, a0.v, false, b.v, (short)0, acc[n], false, false);
            acc[8 + n] = __builtin_amdgcn_wmma_f32_16x16x32_bf16(
                             false, a1.v, false, b.v, (short)0, acc[8 + n], false, false);
        }
    }

    // Elementwise LSTM cell in registers: i,f,g,o tiles live in the same lane slots.
#pragma unroll
    for (int s = 0; s < 2; s++) {
        const int j = j0 + s * 16 + lr;
        const float bi = bias[0 * HH + j];
        const float bf_ = bias[1 * HH + j];
        const float bg = bias[2 * HH + j];
        const float bo = bias[3 * HH + j];
#pragma unroll
        for (int mt = 0; mt < 2; mt++) {
#pragma unroll
            for (int r = 0; r < 8; r++) {
                const int brow = row0 + mt * 16 + half * 8 + r;
                const float gi = acc[mt * 8 + 0 + s][r] + bi;
                const float gf = acc[mt * 8 + 2 + s][r] + bf_;
                const float gg = acc[mt * 8 + 4 + s][r] + bg;
                const float go = acc[mt * 8 + 6 + s][r] + bo;
                const size_t ci = (size_t)brow * HH + j;
                const float cold = c[ci];
                const float cnew = sigf(gf) * cold + sigf(gi) * tanhf(gg);
                const float hnew = sigf(go) * tanhf(cnew);
                c[ci] = cnew;
                xh_next[(size_t)brow * XS + EPAD + j] = f2bf(hnew);
            }
        }
    }
}

// ---------------------------------------------------------------------------
// Per-step output projection: scores_t = h_new @ W_out^T + b_out  (B x 96)
// grid (B/128), 128 threads; wave: 32 rows x 96 cols = 12 WMMA tiles.
// ---------------------------------------------------------------------------
__global__ __launch_bounds__(128) void proj(
        const unsigned short* __restrict__ xh,           // h at column offset EPAD
        const unsigned short* __restrict__ Wout,         // (V, H) bf16
        const float* __restrict__ b_out,
        float* __restrict__ out) {                       // d_out + t*B*V
    const int wave = threadIdx.x >> 5;
    const int lane = threadIdx.x & 31;
    const int half = lane >> 4;
    const int lr   = lane & 15;
    const int row0 = blockIdx.x * 128 + wave * 32;

    v8f acc[12] = {};                                    // [mt*6 + n]

    const unsigned short* arow0 = xh + (size_t)(row0 + lr) * XS + EPAD;
    const unsigned short* arow1 = arow0 + (size_t)16 * XS;

    for (int k = 0; k < HH; k += 32) {
        AFrag a0, a1;
        a0.q[0] = *(const u32x4*)(arow0 + k + half * 8);
        a0.q[1] = *(const u32x4*)(arow0 + k + 16 + half * 8);
        a1.q[0] = *(const u32x4*)(arow1 + k + half * 8);
        a1.q[1] = *(const u32x4*)(arow1 + k + 16 + half * 8);
#pragma unroll
        for (int n = 0; n < 6; n++) {
            const unsigned short* brow =
                Wout + (size_t)(n * 16 + lr) * HH + k + half * 8;
            AFrag b;
            b.q[0] = *(const u32x4*)(brow);
            b.q[1] = *(const u32x4*)(brow + 16);
            acc[n]     = __builtin_amdgcn_wmma_f32_16x16x32_bf16(
                             false, a0.v, false, b.v, (short)0, acc[n], false, false);
            acc[6 + n] = __builtin_amdgcn_wmma_f32_16x16x32_bf16(
                             false, a1.v, false, b.v, (short)0, acc[6 + n], false, false);
        }
    }
#pragma unroll
    for (int n = 0; n < 6; n++) {
        const int v = n * 16 + lr;
        const float bo = b_out[v];
#pragma unroll
        for (int mt = 0; mt < 2; mt++) {
#pragma unroll
            for (int r = 0; r < 8; r++) {
                const int b = row0 + mt * 16 + half * 8 + r;
                out[(size_t)b * VV + v] = acc[mt * 6 + n][r] + bo;
            }
        }
    }
}

// ---------------------------------------------------------------------------
// Host-side launch (graph-capture safe: only kernel launches on `stream`)
// ---------------------------------------------------------------------------
extern "C" void kernel_launch(void* const* d_in, const int* in_sizes, int n_in,
                              void* d_out, int out_size, void* d_ws, size_t ws_size,
                              hipStream_t stream) {
    const int*   ids  = (const int*)  d_in[0];   // (L, B)
    const float* h0   = (const float*)d_in[1];   // (1, B, H)
    const float* c0   = (const float*)d_in[2];   // (1, B, H)
    const float* emb  = (const float*)d_in[3];   // (V, E)
    const float* Wih  = (const float*)d_in[4];   // (4H, E)
    const float* Whh  = (const float*)d_in[5];   // (4H, H)
    const float* bih  = (const float*)d_in[6];   // (4H,)
    const float* bhh  = (const float*)d_in[7];   // (4H,)
    const float* WoutF= (const float*)d_in[8];   // (V, H)
    const float* bout = (const float*)d_in[9];   // (V,)
    float* out = (float*)d_out;                  // scores (L,B,V) ++ hL (B,H) ++ cL (B,H)

    // workspace carve (256 B aligned)
    uint8_t* w = (uint8_t*)d_ws;
    auto carve = [&](size_t bytes) {
        uint8_t* p = w;
        w += (bytes + 255) & ~(size_t)255;
        return p;
    };
    unsigned short* Wcat = (unsigned short*)carve((size_t)4 * HH * KT * 2);  // 2.25 MB
    unsigned short* embp = (unsigned short*)carve((size_t)VV * EPAD * 2);
    unsigned short* Woutb= (unsigned short*)carve((size_t)VV * HH * 2);
    float*          bias = (float*)         carve((size_t)4 * HH * 4);
    unsigned short* xhA  = (unsigned short*)carve((size_t)BB * XS * 2);      // 2.25 MB
    unsigned short* xhB  = (unsigned short*)carve((size_t)BB * XS * 2);      // 2.25 MB
    float*          cbuf = (float*)         carve((size_t)BB * HH * 4);      // 4 MB

    // one-time packs
    pack_wcat<<<(4 * HH * KT + 255) / 256, 256, 0, stream>>>(Wih, Whh, Wcat);
    pack_emb <<<(VV * EPAD + 255) / 256, 256, 0, stream>>>(emb, embp);
    pack_wout<<<(VV * HH + 255) / 256, 256, 0, stream>>>(WoutF, Woutb);
    pack_bias<<<(4 * HH + 255) / 256, 256, 0, stream>>>(bih, bhh, bias);
    init_state<<<(BB * HH + 255) / 256, 256, 0, stream>>>(h0, c0, xhA, cbuf);

    // sequential recurrence: ping-pong [x|h] buffers
    for (int t = 0; t < LL; t++) {
        unsigned short* xc = (t & 1) ? xhB : xhA;
        unsigned short* xn = (t & 1) ? xhA : xhB;
        embed_x<<<(BB * EPAD + 255) / 256, 256, 0, stream>>>(ids + (size_t)t * BB, embp, xc);
        lstm_step<<<dim3(BB / 128, HH / 32), 128, 0, stream>>>(xc, xn, Wcat, bias, cbuf);
        proj<<<BB / 128, 128, 0, stream>>>(xn, Woutb, bout, out + (size_t)t * BB * VV);
    }

    // hL, cL (after t=63, h lives in xhA)
    finalize<<<(BB * HH + 255) / 256, 256, 0, stream>>>(xhA, cbuf,
                                                        out + (size_t)LL * BB * VV);
}
